// GNNens_28741921145078
// MI455X (gfx1250) — compile-verified, matrix operand order checked by reading, not run on hardware
//
#include <hip/hip_runtime.h>
#include <stdint.h>

typedef __attribute__((ext_vector_type(16))) _Float16 v16h;
typedef __attribute__((ext_vector_type(8)))  _Float16 v8h;
typedef __attribute__((ext_vector_type(8)))  float    v8f;

#define LAT 128
#define OUT 512
#define NCOMP 16
#define MTILE 32
#define LAT_LD 136   // 128 + 8 pad: A-frag rows hit 64 distinct LDS banks
#define HID_LD 520   // 512 + 8 pad: same property, rows 16B aligned

// Async copy global->LDS, 16B/lane (512B per wave op), ASYNCcnt-tracked.
// INST_OFFSET is added to BOTH the LDS and global addresses (ISA 08 §4.4),
// so a linear region copy needs only one base per side + immediate offsets.
#define ASYNC_STAGE_B128(ldsAddr, gAddr, imm)                                  \
    asm volatile("global_load_async_to_lds_b128 %0, %1, off offset:%2"         \
                 :: "v"(ldsAddr), "v"(gAddr), "n"(imm) : "memory")
#define WAIT_ASYNC_8() asm volatile("s_wait_asynccnt 0x8" ::: "memory")
#define WAIT_ASYNC_0() asm volatile("s_wait_asynccnt 0x0" ::: "memory")

__device__ inline uint32_t lds_addr32(const void* p) {
    return (uint32_t)(uintptr_t)p;   // low 32 bits of generic addr = LDS offset
}

// ---- fragment loaders -------------------------------------------------------
// A-fragment (16x32 f16) from LDS, row-major, stride ld (ISA 7.12.2 layout).
__device__ inline v16h frag_lds(const _Float16* base, int ld, int lane, int m0, int k0) {
    const int r = m0 + (lane & 15);
    const int g = (lane >> 4) & 1;
    const _Float16* p = base + r * ld + k0 + g * 8;
    v8h lo = *(const v8h*)(p);
    v8h hi = *(const v8h*)(p + 16);
    return __builtin_shufflevector(lo, hi, 0,1,2,3,4,5,6,7,8,9,10,11,12,13,14,15);
}

// B-fragment from an LDS-staged split-half region: per col-tile ctl, 512B of
// per-lane low halves then 512B of high halves (two linear conflict-free b128s).
__device__ inline v16h frag_b_staged(const _Float16* buf, int ctl, int lane) {
    const _Float16* p = buf + ctl * 512 + lane * 8;   // ctl*1024B + lane*16B
    v8h lo = *(const v8h*)(p);
    v8h hi = *(const v8h*)(p + 256);                  // +512B
    return __builtin_shufflevector(lo, hi, 0,1,2,3,4,5,6,7,8,9,10,11,12,13,14,15);
}

// Fallback B-fragment from row-major fp32 W[n,k] (B^T), inline cvt.
__device__ inline v16h frag_b_f32(const float* base, int ld, int lane, int n0, int k0) {
    const int r = n0 + (lane & 15);
    const int g = (lane >> 4) & 1;
    const float* p = base + (size_t)r * ld + k0 + g * 8;
    float4 a0 = *(const float4*)(p + 0);
    float4 a1 = *(const float4*)(p + 4);
    float4 b0 = *(const float4*)(p + 16);
    float4 b1 = *(const float4*)(p + 20);
    v16h v;
    v[0]=(_Float16)a0.x; v[1]=(_Float16)a0.y; v[2]=(_Float16)a0.z; v[3]=(_Float16)a0.w;
    v[4]=(_Float16)a1.x; v[5]=(_Float16)a1.y; v[6]=(_Float16)a1.z; v[7]=(_Float16)a1.w;
    v[8]=(_Float16)b0.x; v[9]=(_Float16)b0.y; v[10]=(_Float16)b0.z; v[11]=(_Float16)b0.w;
    v[12]=(_Float16)b1.x; v[13]=(_Float16)b1.y; v[14]=(_Float16)b1.z; v[15]=(_Float16)b1.w;
    return v;
}

// ---- weight swizzle: fp32 (O x K row-major per expert) -> f16, split-half
// fragment order. Region (e,kc) = 32KB: nt*1024B + {0,512}B + lane*16B.
__global__ void k_swizzle(const float* __restrict__ W, _Float16* __restrict__ Wsw,
                          int K, int KC) {
    const int idx  = blockIdx.x * 256 + threadIdx.x;   // (e*KC+kc, nt, lane) flat
    const int lane = idx & 31;
    const int nt   = (idx >> 5) & 31;
    const int rest = idx >> 10;                        // e*KC + kc
    const int kc   = rest % KC;
    const int e    = rest / KC;
    const int g    = (lane >> 4) & 1;
    const int row  = nt * 16 + (lane & 15);
    const float* src = W + ((size_t)e * OUT + row) * K + kc * 32 + g * 8;
    float4 a0 = *(const float4*)(src + 0);
    float4 a1 = *(const float4*)(src + 4);
    float4 b0 = *(const float4*)(src + 16);
    float4 b1 = *(const float4*)(src + 20);
    v8h lo, hi;
    lo[0]=(_Float16)a0.x; lo[1]=(_Float16)a0.y; lo[2]=(_Float16)a0.z; lo[3]=(_Float16)a0.w;
    lo[4]=(_Float16)a1.x; lo[5]=(_Float16)a1.y; lo[6]=(_Float16)a1.z; lo[7]=(_Float16)a1.w;
    hi[0]=(_Float16)b0.x; hi[1]=(_Float16)b0.y; hi[2]=(_Float16)b0.z; hi[3]=(_Float16)b0.w;
    hi[4]=(_Float16)b1.x; hi[5]=(_Float16)b1.y; hi[6]=(_Float16)b1.z; hi[7]=(_Float16)b1.w;
    _Float16* dst = Wsw + (size_t)rest * 16384 + nt * 512 + lane * 8;
    *(v8h*)(dst)       = lo;
    *(v8h*)(dst + 256) = hi;
}

// ---- setup kernels ----------------------------------------------------------
__global__ void k_init(int* counts) {
    if (threadIdx.x < NCOMP) counts[threadIdx.x] = 0;
}

__global__ void k_hist(const int* __restrict__ c, int n, int* counts) {
    int i = blockIdx.x * 256 + threadIdx.x;
    if (i < n) atomicAdd(&counts[c[i]], 1);
}

__global__ void k_scan(const int* __restrict__ counts, int* offsets, int* cursor,
                       int* totalTiles, int* tile_e, int* tile_r) {
    if (threadIdx.x != 0) return;
    int off = 0, t = 0;
    for (int e = 0; e < NCOMP; ++e) {
        offsets[e] = off;
        cursor[e]  = off;
        int cnt = counts[e];
        for (int s = 0; s < cnt; s += MTILE) { tile_e[t] = e; tile_r[t] = off + s; ++t; }
        off += cnt;
    }
    offsets[NCOMP] = off;
    *totalTiles = t;
}

__global__ void k_scatter(const int* __restrict__ c, int n, int* cursor,
                          int* __restrict__ sorted) {
    int i = blockIdx.x * 256 + threadIdx.x;
    if (i < n) {
        int p = atomicAdd(&cursor[c[i]], 1);
        sorted[p] = i;
    }
}

// Stage one (e,kc) wave-region (4KB) into a per-wave LDS buffer: 8 async b128
// ops with immediate offsets (applied to both LDS and global sides).
__device__ inline void stage_region(uint32_t ldsBase, const _Float16* gBase) {
    const uint64_t g = (uint64_t)(uintptr_t)gBase;
    ASYNC_STAGE_B128(ldsBase, g, 0);
    ASYNC_STAGE_B128(ldsBase, g, 512);
    ASYNC_STAGE_B128(ldsBase, g, 1024);
    ASYNC_STAGE_B128(ldsBase, g, 1536);
    ASYNC_STAGE_B128(ldsBase, g, 2048);
    ASYNC_STAGE_B128(ldsBase, g, 2560);
    ASYNC_STAGE_B128(ldsBase, g, 3072);
    ASYNC_STAGE_B128(ldsBase, g, 3584);
}

// ---- main grouped-GEMM kernel ----------------------------------------------
// One block = 32 rows of one expert; 8 waves; wave w owns 64 output columns
// x all 32 rows (8 v8f accumulators). F16W path: B fragments staged
// global->LDS with async-DMA, double buffered one k-chunk ahead.
template <bool F16W>
__global__ __launch_bounds__(256) void k_moe_gemm(
    const float* __restrict__ eps, const float* __restrict__ mu,
    const float* __restrict__ sigma, const float* __restrict__ W1,
    const float* __restrict__ b1, const float* __restrict__ W2,
    const float* __restrict__ b2,
    const _Float16* __restrict__ W1h, const _Float16* __restrict__ W2h,
    float* __restrict__ out,
    const int* __restrict__ offsets, const int* __restrict__ sorted,
    const int* __restrict__ totalTiles, const int* __restrict__ tile_e,
    const int* __restrict__ tile_r)
{
    __shared__ __align__(16) _Float16 latS[MTILE * LAT_LD];
    __shared__ __align__(16) _Float16 hidS[MTILE * HID_LD];
    __shared__ __align__(16) _Float16 bstage[8][2][2048];   // per-wave 2x4KB

    const int t = blockIdx.x;
    if (t >= *totalTiles) return;

    const int e        = tile_e[t];
    const int rowstart = tile_r[t];
    const int rowend   = offsets[e + 1];

    const int tid  = threadIdx.x;
    const int lane = tid & 31;
    const int w    = tid >> 5;                // wave id: cols [w*64, w*64+64)
    const int hi8  = (lane >> 4) << 3;

    // Per-wave staging state (F16W path): lane*16B folded into the bases.
    const uint32_t lbuf0 = lds_addr32(&bstage[w][0][0]) + (uint32_t)lane * 16;
    const uint32_t lbuf1 = lds_addr32(&bstage[w][1][0]) + (uint32_t)lane * 16;
    const _Float16* g1 = W1h + ((size_t)e * 4)  * 16384 + w * 2048 + lane * 8;
    const _Float16* g2 = W2h + ((size_t)e * 16) * 16384 + w * 2048 + lane * 8;

    // ---- layer-1 kc=0 B stage: issue before staging/barrier ----------------
    if (F16W) stage_region(lbuf0, g1);

    // ---- stage lat = eps*sigma[e] + mu[e] as f16 into LDS (32 x 128) -------
    {
        const int lr = tid >> 3;
        const int cg = (tid & 7) * 16;
        const int gr = rowstart + lr;
        const int cl = (gr < rowend) ? gr : (rowend - 1);
        const int sr = sorted[cl];
        const float4* e4 = (const float4*)(eps + (size_t)sr * LAT + cg);
        const float4* m4 = (const float4*)(mu    + e * LAT + cg);
        const float4* s4 = (const float4*)(sigma + e * LAT + cg);
        _Float16* dst = &latS[lr * LAT_LD + cg];
        #pragma unroll
        for (int q = 0; q < 4; ++q) {
            float4 ev = e4[q], mv = m4[q], sv = s4[q];
            dst[q*4+0] = (_Float16)(ev.x * sv.x + mv.x);
            dst[q*4+1] = (_Float16)(ev.y * sv.y + mv.y);
            dst[q*4+2] = (_Float16)(ev.z * sv.z + mv.z);
            dst[q*4+3] = (_Float16)(ev.w * sv.w + mv.w);
        }
    }
    __syncthreads();

    // ---- layer 1: hidden = relu(lat @ W1[e]^T + b1[e]) ---------------------
    const float* W1e = W1 + (size_t)e * OUT * LAT;
    const float* b1e = b1 + e * OUT;
    {
        v8f acc[8];                            // [rb*4 + ctl]
        #pragma unroll
        for (int ctl = 0; ctl < 4; ++ctl) {
            const float bv = b1e[w * 64 + ctl * 16 + (lane & 15)];
            #pragma unroll
            for (int j = 0; j < 8; ++j) { acc[ctl][j] = bv; acc[4 + ctl][j] = bv; }
        }
        #pragma unroll
        for (int kc = 0; kc < 4; ++kc) {
            if (F16W && kc + 1 < 4)
                stage_region((kc + 1) & 1 ? lbuf1 : lbuf0, g1 + (kc + 1) * 16384);
            v16h a0 = frag_lds(latS, LAT_LD, lane, 0,  kc * 32);
            v16h a1 = frag_lds(latS, LAT_LD, lane, 16, kc * 32);
            if (F16W) { if (kc + 1 < 4) WAIT_ASYNC_8(); else WAIT_ASYNC_0(); }
            const _Float16* buf = &bstage[w][kc & 1][0];
            v16h bf[4];
            #pragma unroll
            for (int ctl = 0; ctl < 4; ++ctl)
                bf[ctl] = F16W ? frag_b_staged(buf, ctl, lane)
                               : frag_b_f32(W1e, LAT, lane, (w * 4 + ctl) * 16, kc * 32);
            #pragma unroll
            for (int ctl = 0; ctl < 4; ++ctl) {
                acc[ctl] = __builtin_amdgcn_wmma_f32_16x16x32_f16(
                    false, a0, false, bf[ctl], (short)0, acc[ctl], false, false);
                acc[4 + ctl] = __builtin_amdgcn_wmma_f32_16x16x32_f16(
                    false, a1, false, bf[ctl], (short)0, acc[4 + ctl], false, false);
            }
        }
        #pragma unroll
        for (int rb = 0; rb < 2; ++rb) {
            #pragma unroll
            for (int ctl = 0; ctl < 4; ++ctl) {
                #pragma unroll
                for (int j = 0; j < 8; ++j) {
                    float v = acc[rb * 4 + ctl][j];
                    v = v > 0.0f ? v : 0.0f;                 // relu
                    const int m = rb * 16 + j + hi8;
                    hidS[m * HID_LD + w * 64 + ctl * 16 + (lane & 15)] = (_Float16)v;
                }
            }
        }
    }

    // ---- layer-2 kc=0 B stage: issue before the barrier --------------------
    if (F16W) stage_region(lbuf0, g2);
    __syncthreads();

    // ---- layer 2: out = hidden @ W2[e]^T + b2[e] ---------------------------
    const float* W2e = W2 + (size_t)e * OUT * OUT;
    const float* b2e = b2 + e * OUT;

    int  srow[2][8];
    bool valid[2][8];
    #pragma unroll
    for (int rb = 0; rb < 2; ++rb) {
        #pragma unroll
        for (int j = 0; j < 8; ++j) {
            const int gr = rowstart + rb * 16 + j + hi8;
            valid[rb][j] = (gr < rowend);
            srow[rb][j]  = sorted[valid[rb][j] ? gr : (rowend - 1)];
        }
    }

    {
        v8f acc[8];
        #pragma unroll
        for (int ctl = 0; ctl < 4; ++ctl) {
            const float bv = b2e[w * 64 + ctl * 16 + (lane & 15)];
            #pragma unroll
            for (int j = 0; j < 8; ++j) { acc[ctl][j] = bv; acc[4 + ctl][j] = bv; }
        }
        #pragma unroll
        for (int kc = 0; kc < 16; ++kc) {
            if (F16W && kc + 1 < 16)
                stage_region((kc + 1) & 1 ? lbuf1 : lbuf0, g2 + (kc + 1) * 16384);
            v16h a0 = frag_lds(hidS, HID_LD, lane, 0,  kc * 32);
            v16h a1 = frag_lds(hidS, HID_LD, lane, 16, kc * 32);
            if (F16W) { if (kc + 1 < 16) WAIT_ASYNC_8(); else WAIT_ASYNC_0(); }
            const _Float16* buf = &bstage[w][kc & 1][0];
            v16h bf[4];
            #pragma unroll
            for (int ctl = 0; ctl < 4; ++ctl)
                bf[ctl] = F16W ? frag_b_staged(buf, ctl, lane)
                               : frag_b_f32(W2e, OUT, lane, (w * 4 + ctl) * 16, kc * 32);
            #pragma unroll
            for (int ctl = 0; ctl < 4; ++ctl) {
                acc[ctl] = __builtin_amdgcn_wmma_f32_16x16x32_f16(
                    false, a0, false, bf[ctl], (short)0, acc[ctl], false, false);
                acc[4 + ctl] = __builtin_amdgcn_wmma_f32_16x16x32_f16(
                    false, a1, false, bf[ctl], (short)0, acc[4 + ctl], false, false);
            }
        }
        #pragma unroll
        for (int rb = 0; rb < 2; ++rb) {
            #pragma unroll
            for (int ctl = 0; ctl < 4; ++ctl) {
                #pragma unroll
                for (int j = 0; j < 8; ++j) {
                    if (valid[rb][j])
                        out[(size_t)srow[rb][j] * OUT + w * 64 + ctl * 16 + (lane & 15)] =
                            acc[rb * 4 + ctl][j];
                }
            }
        }
    }
}

// ---- host launcher ----------------------------------------------------------
extern "C" void kernel_launch(void* const* d_in, const int* in_sizes, int n_in,
                              void* d_out, int out_size, void* d_ws, size_t ws_size,
                              hipStream_t stream) {
    const float* eps   = (const float*)d_in[1];
    const int*   c     = (const int*)  d_in[2];
    const float* mu    = (const float*)d_in[3];
    const float* sigma = (const float*)d_in[4];
    const float* W1    = (const float*)d_in[5];
    const float* b1    = (const float*)d_in[6];
    const float* W2    = (const float*)d_in[7];
    const float* b2    = (const float*)d_in[8];
    float* out = (float*)d_out;

    const int n    = in_sizes[1] / LAT;                 // eps is (n, 128)
    const int maxT = (n + MTILE - 1) / MTILE + NCOMP;   // upper bound on tiles

    int* counts  = (int*)d_ws;           // 16
    int* offsets = counts + NCOMP;       // 17
    int* cursor  = offsets + NCOMP + 1;  // 16
    int* totalT  = cursor + NCOMP;       // 1
    int* tile_e  = totalT + 1;           // maxT
    int* tile_r  = tile_e + maxT;        // maxT
    int* sorted  = tile_r + maxT;        // n

    const size_t sortBytes = (size_t)(50 + 2 * maxT + n) * sizeof(int);
    const size_t wOff    = (sortBytes + 255) & ~(size_t)255;
    const size_t w1Elems = (size_t)NCOMP * OUT * LAT;   // 2 MB f16
    const size_t w2Elems = (size_t)NCOMP * OUT * OUT;   // 8 MB f16
    const size_t need    = wOff + (w1Elems + w2Elems) * sizeof(_Float16);
    const bool useF16W   = (ws_size >= need);
    _Float16* W1h = (_Float16*)((char*)d_ws + wOff);
    _Float16* W2h = W1h + w1Elems;

    const int nb = (n + 255) / 256;
    k_init   <<<1, 64, 0, stream>>>(counts);
    k_hist   <<<nb, 256, 0, stream>>>(c, n, counts);
    k_scan   <<<1, 32, 0, stream>>>(counts, offsets, cursor, totalT, tile_e, tile_r);
    k_scatter<<<nb, 256, 0, stream>>>(c, n, cursor, sorted);

    if (useF16W) {
        k_swizzle<<<(NCOMP * 4  * 1024) / 256, 256, 0, stream>>>(W1, W1h, LAT, 4);
        k_swizzle<<<(NCOMP * 16 * 1024) / 256, 256, 0, stream>>>(W2, W2h, OUT, 16);
        k_moe_gemm<true><<<maxT, 256, 0, stream>>>(eps, mu, sigma, W1, b1, W2, b2,
                                                   W1h, W2h, out, offsets, sorted,
                                                   totalT, tile_e, tile_r);
    } else {
        k_moe_gemm<false><<<maxT, 256, 0, stream>>>(eps, mu, sigma, W1, b1, W2, b2,
                                                    W1h, W2h, out, offsets, sorted,
                                                    totalT, tile_e, tile_r);
    }
}
